// FRC_79190607004340
// MI455X (gfx1250) — matrix operand order repaired and checked by reading, compiled, not run on hardware
//
#include <hip/hip_runtime.h>
#include <hip/hip_bf16.h>
#include <stdint.h>

typedef __attribute__((ext_vector_type(16))) __bf16 v16bf;
typedef __attribute__((ext_vector_type(8)))  float  v8f;
typedef __attribute__((ext_vector_type(4)))  __bf16 v4bf;

#define ROWS 64
#define DDIM 1024
#define KCL  64
#define THREADS 256

// dynamic LDS byte offsets
#define OFF_XROW   0            // ROWS*DDIM*2  = 131072
#define OFF_XT     131072       // DDIM*ROWS*2  = 131072
#define OFF_LOGITS 262144       // ROWS*KCL*4   = 16384
#define OFF_AT     278528       // KCL*ROWS*2   = 8192
#define SMEM_BYTES 286720
#define OFF_RED    262144       // aliases logits region (used before GEMM1 output)
#define OFF_INV    263168

union FragU { uint4 q[2]; v16bf v; };

__global__ void __launch_bounds__(256) zero_ws_kernel(float* __restrict__ p, int n) {
    int i = blockIdx.x * blockDim.x + threadIdx.x;
    if (i < n) p[i] = 0.0f;
}

__global__ void __launch_bounds__(256) cvt_w_kernel(const float* __restrict__ W,
                                                    __bf16* __restrict__ Wb, int n) {
    int i = blockIdx.x * blockDim.x + threadIdx.x;
    if (i < n) Wb[i] = (__bf16)W[i];
}

__global__ void __launch_bounds__(THREADS)
netvlad_main(const float* __restrict__ grids,
             const __bf16* __restrict__ Wb,
             const float* __restrict__ bias,
             float* __restrict__ acc_p,
             float* __restrict__ asum)
{
    extern __shared__ char smem[];
    __bf16* xrow = (__bf16*)(smem + OFF_XROW);   // [ROWS][DDIM]
    __bf16* xT   = (__bf16*)(smem + OFF_XT);     // [DDIM][ROWS]
    float*  lgt  = (float*) (smem + OFF_LOGITS); // [ROWS][KCL]
    __bf16* aT   = (__bf16*)(smem + OFF_AT);     // [KCL][ROWS]
    float*  red  = (float*) (smem + OFF_RED);    // [256]
    float*  invn = (float*) (smem + OFF_INV);    // [ROWS]

    const int tid  = threadIdx.x;
    const int lane = tid & 31;
    const int wave = tid >> 5;
    const int half = lane >> 4;
    const int l16  = lane & 15;

    const int bi    = blockIdx.x >> 5;   // batch index (32 chunks per batch)
    const int chunk = blockIdx.x & 31;   // 64-row chunk within batch
    const float* gbase = grids + ((size_t)bi * 2048 + (size_t)chunk * ROWS) * DDIM;

    // ---------- Phase 1a: per-row sum of squares (4 threads per row) ----------
    const int rn = tid >> 2;    // row 0..63
    const int qq = tid & 3;     // quarter of D
    const float4* g4 = (const float4*)(gbase + (size_t)rn * DDIM + qq * 256);
    float ss = 0.0f;
    #pragma unroll 8
    for (int j = 0; j < 64; ++j) {
        float4 v = g4[j];
        ss += v.x * v.x + v.y * v.y + v.z * v.z + v.w * v.w;
    }
    red[tid] = ss;
    __syncthreads();
    if (tid < ROWS) {
        float s = red[tid * 4 + 0] + red[tid * 4 + 1] + red[tid * 4 + 2] + red[tid * 4 + 3];
        invn[tid] = 1.0f / fmaxf(sqrtf(s), 1e-12f);
    }
    __syncthreads();

    // ---------- Phase 1b: normalize -> bf16, write xrow (row-major) and xT ----------
    {
        float inv = invn[rn];
        #pragma unroll 4
        for (int j = 0; j < 64; ++j) {
            float4 v = g4[j];
            __bf16 h0 = (__bf16)(v.x * inv);
            __bf16 h1 = (__bf16)(v.y * inv);
            __bf16 h2 = (__bf16)(v.z * inv);
            __bf16 h3 = (__bf16)(v.w * inv);
            int d0 = qq * 256 + j * 4;
            v4bf pk = {h0, h1, h2, h3};
            *(v4bf*)(xrow + rn * DDIM + d0) = pk;
            xT[(d0 + 0) * ROWS + rn] = h0;
            xT[(d0 + 1) * ROWS + rn] = h1;
            xT[(d0 + 2) * ROWS + rn] = h2;
            xT[(d0 + 3) * ROWS + rn] = h3;
        }
    }
    __syncthreads();

    // ---------- GEMM1: logits[ROWS x K] = xrow * Wb^T  (WMMA bf16, Kdim = d) ----------
    {
        const int mt  = wave >> 1;          // row tile (0..3)
        const int nt0 = (wave & 1) * 2;     // first of two K tiles
        v8f c0 = {};
        v8f c1 = {};
        const int arow = mt * 16 + l16;
        const __bf16* arp = xrow + (size_t)arow * DDIM;
        const __bf16* b0p = Wb + (size_t)(nt0 * 16 + l16) * DDIM;
        const __bf16* b1p = Wb + (size_t)((nt0 + 1) * 16 + l16) * DDIM;
        for (int kt = 0; kt < 32; ++kt) {
            const int dbase = kt * 32;
            FragU a, b0, b1;
            a.q[0]  = *(const uint4*)(arp + dbase + 16 * half);
            a.q[1]  = *(const uint4*)(arp + dbase + 16 * half + 8);
            const uint4* p0 = (const uint4*)(b0p + dbase + 16 * half);
            b0.q[0] = p0[0]; b0.q[1] = p0[1];
            const uint4* p1 = (const uint4*)(b1p + dbase + 16 * half);
            b1.q[0] = p1[0]; b1.q[1] = p1[1];
            c0 = __builtin_amdgcn_wmma_f32_16x16x32_bf16(false, a.v, false, b0.v,
                                                         (short)0, c0, false, false);
            c1 = __builtin_amdgcn_wmma_f32_16x16x32_bf16(false, a.v, false, b1.v,
                                                         (short)0, c1, false, false);
        }
        #pragma unroll
        for (int g = 0; g < 8; ++g) {
            int n_loc = mt * 16 + g + 8 * half;
            lgt[n_loc * KCL + nt0 * 16 + l16]       = c0[g];
            lgt[n_loc * KCL + (nt0 + 1) * 16 + l16] = c1[g];
        }
    }
    __syncthreads();

    // ---------- Softmax over K per row; write aT (transposed, bf16) ----------
    if (tid < ROWS) {
        const int n = tid;
        float m = -3.4e38f;
        for (int kk = 0; kk < KCL; ++kk) {
            float l = lgt[n * KCL + kk] + bias[kk];
            lgt[n * KCL + kk] = l;
            m = fmaxf(m, l);
        }
        float s = 0.0f;
        for (int kk = 0; kk < KCL; ++kk) {
            float e = __expf(lgt[n * KCL + kk] - m);
            lgt[n * KCL + kk] = e;
            s += e;
        }
        float is = 1.0f / s;
        for (int kk = 0; kk < KCL; ++kk)
            aT[kk * ROWS + n] = (__bf16)(lgt[n * KCL + kk] * is);
    }
    __syncthreads();

    // ---------- asum partial: one thread per cluster ----------
    if (tid < KCL) {
        const int kk = tid;
        float s = 0.0f;
        for (int n = 0; n < ROWS; ++n) s += (float)aT[kk * ROWS + n];
        atomicAdd(&asum[bi * KCL + kk], s);
    }

    // ---------- GEMM2: p_partial[K x D] += aT * x  (WMMA bf16, Kdim = n) ----------
    {
        // Preload all A fragments (aT): 4 K-tiles x 2 n-steps
        FragU afr[4][2];
        #pragma unroll
        for (int mt2 = 0; mt2 < 4; ++mt2) {
            const __bf16* ar = aT + (size_t)(mt2 * 16 + l16) * ROWS;
            #pragma unroll
            for (int ks = 0; ks < 2; ++ks) {
                int nb = ks * 32;
                afr[mt2][ks].q[0] = *(const uint4*)(ar + nb + 16 * half);
                afr[mt2][ks].q[1] = *(const uint4*)(ar + nb + 16 * half + 8);
            }
        }
        float* pb = acc_p + (size_t)bi * KCL * DDIM;
        for (int j = 0; j < 8; ++j) {
            const int nt2  = wave * 8 + j;        // d tile 0..63
            const int dcol = nt2 * 16 + l16;
            v8f a0 = {}; v8f a1 = {}; v8f a2 = {}; v8f a3 = {};
            #pragma unroll
            for (int ks = 0; ks < 2; ++ks) {
                int nb = ks * 32;
                FragU b;
                const uint4* bp = (const uint4*)(xT + (size_t)dcol * ROWS + nb + 16 * half);
                b.q[0] = bp[0]; b.q[1] = bp[1];
                a0 = __builtin_amdgcn_wmma_f32_16x16x32_bf16(false, afr[0][ks].v, false, b.v,
                                                             (short)0, a0, false, false);
                a1 = __builtin_amdgcn_wmma_f32_16x16x32_bf16(false, afr[1][ks].v, false, b.v,
                                                             (short)0, a1, false, false);
                a2 = __builtin_amdgcn_wmma_f32_16x16x32_bf16(false, afr[2][ks].v, false, b.v,
                                                             (short)0, a2, false, false);
                a3 = __builtin_amdgcn_wmma_f32_16x16x32_bf16(false, afr[3][ks].v, false, b.v,
                                                             (short)0, a3, false, false);
            }
            #pragma unroll
            for (int g = 0; g < 8; ++g) {
                int r = g + 8 * half;
                atomicAdd(pb + (size_t)(r)      * DDIM + dcol, a0[g]);
                atomicAdd(pb + (size_t)(16 + r) * DDIM + dcol, a1[g]);
                atomicAdd(pb + (size_t)(32 + r) * DDIM + dcol, a2[g]);
                atomicAdd(pb + (size_t)(48 + r) * DDIM + dcol, a3[g]);
            }
        }
    }
}

__global__ void __launch_bounds__(256)
netvlad_finalize(const float* __restrict__ acc_p, const float* __restrict__ asum,
                 const float* __restrict__ centroids, float* __restrict__ out)
{
    __shared__ float sred[256];
    const int row = blockIdx.x;        // b*K + kk
    const int kk  = row & 63;
    const float* ap = acc_p + (size_t)row * DDIM;
    const float* cp = centroids + (size_t)kk * DDIM;
    const float asv = asum[row];
    float v[4];
    float ss = 0.0f;
    #pragma unroll
    for (int r = 0; r < 4; ++r) {
        int d = threadIdx.x + r * 256;
        float t = ap[d] - cp[d] * asv;
        v[r] = t;
        ss += t * t;
    }
    sred[threadIdx.x] = ss;
    __syncthreads();
    for (int s = 128; s > 0; s >>= 1) {
        if (threadIdx.x < s) sred[threadIdx.x] += sred[threadIdx.x + s];
        __syncthreads();
    }
    float inv = 1.0f / fmaxf(sqrtf(sred[0]), 1e-12f);
    #pragma unroll
    for (int r = 0; r < 4; ++r)
        out[(size_t)row * DDIM + threadIdx.x + r * 256] = v[r] * inv;
}

extern "C" void kernel_launch(void* const* d_in, const int* in_sizes, int n_in,
                              void* d_out, int out_size, void* d_ws, size_t ws_size,
                              hipStream_t stream)
{
    // inputs: idx, grids[16,2048,1024] f32, W[64,1024] f32, b[64] f32, centroids[64,1024] f32
    const float* grids     = (const float*)d_in[1];
    const float* W         = (const float*)d_in[2];
    const float* bias      = (const float*)d_in[3];
    const float* centroids = (const float*)d_in[4];
    float* out = (float*)d_out;

    char* ws = (char*)d_ws;
    float*  acc_p = (float*)ws;                                   // 16*64*1024 f32 = 4 MB
    float*  asum  = (float*)(ws + 4u * 1024u * 1024u);            // 1024 f32 (contiguous after acc_p)
    __bf16* Wb    = (__bf16*)(ws + 4u * 1024u * 1024u + 4096u);   // 64*1024 bf16 = 128 KB

    const int nzero = 16 * 64 * 1024 + 1024;   // acc_p + asum (contiguous)
    zero_ws_kernel<<<(nzero + 255) / 256, 256, 0, stream>>>(acc_p, nzero);
    cvt_w_kernel<<<(65536 + 255) / 256, 256, 0, stream>>>(W, Wb, 65536);
    netvlad_main<<<512, THREADS, SMEM_BYTES, stream>>>(grids, Wb, bias, acc_p, asum);
    netvlad_finalize<<<1024, 256, 0, stream>>>(acc_p, asum, centroids, out);
}
